// M2_5514738008541
// MI455X (gfx1250) — matrix operand order repaired and checked by reading, not compiled, not run on hardware
//
#include <hip/hip_runtime.h>
#include <hip/hip_bf16.h>

// ---------------------------------------------------------------------------
// GINEConv x3 + final linear, optimized for MI455X (gfx1250, wave32, WMMA).
//   - Edge MLP fused into gather/relu/scatter-add (memory/atomic bound path).
//   - Node MLP GEMMs via v_wmma_f32_16x16x32_f16 with pre-packed fragments.
//   - N = 50000 (= 3125 * 16 exactly), E = 800000, H = 512.
// ---------------------------------------------------------------------------

typedef __attribute__((ext_vector_type(16))) _Float16 v16h;
typedef __attribute__((ext_vector_type(8)))  float    v8f;

#define GN_N      50000
#define GN_E      800000
#define GN_H      512
#define GN_SLOPE  0.01f
#define EDGE_TILE 16

// ---------------------------------------------------------------------------
// zero fill
__global__ void gine_zero_f32(float* __restrict__ p, long n) {
  long i = (long)blockIdx.x * blockDim.x + threadIdx.x;
  if (i < n) p[i] = 0.0f;
}

// ---------------------------------------------------------------------------
// Fused GINE message kernel:
//   msg = relu(x[src] + edge_attr @ elw + elb);  agg[dst] += msg  (atomics)
// One block handles EDGE_TILE edges; edge_attr tile staged in LDS; the
// 16-wide K of the edge MLP is done as VALU MACs (elw fully L2-resident).
__global__ void __launch_bounds__(256)
gine_edge_fused(const float* __restrict__ x,        // [N, in_ch]
                const int*   __restrict__ ei,       // [2, E]
                const float* __restrict__ ea,       // [E, 16]
                const float* __restrict__ elw,      // [16, in_ch]
                const float* __restrict__ elb,      // [in_ch]
                float*       __restrict__ agg,      // [N, in_ch]
                int in_ch, int E) {
  __shared__ float sEA[EDGE_TILE][16];
  __shared__ int   sSrc[EDGE_TILE];
  __shared__ int   sDst[EDGE_TILE];
  const int e0  = blockIdx.x * EDGE_TILE;
  const int tid = threadIdx.x;

  {  // stage 16 edges x 16 attrs (256 floats) + indices
    int e = e0 + (tid >> 4);
    sEA[tid >> 4][tid & 15] = (e < E) ? ea[(long)e * 16 + (tid & 15)] : 0.0f;
  }
  if (tid < EDGE_TILE) {
    int e = e0 + tid;
    sSrc[tid] = (e < E) ? ei[e] : 0;
    sDst[tid] = (e < E) ? ei[E + e] : 0;
  }
  __syncthreads();

  for (int f = tid; f < in_ch; f += 256) {
    float wcol[16];
#pragma unroll
    for (int k = 0; k < 16; ++k) wcol[k] = elw[k * in_ch + f];
    const float bias = elb[f];
#pragma unroll 4
    for (int e = 0; e < EDGE_TILE; ++e) {
      if (e0 + e >= E) break;
      float v = bias;
#pragma unroll
      for (int k = 0; k < 16; ++k) v = fmaf(sEA[e][k], wcol[k], v);
      v += x[(long)sSrc[e] * in_ch + f];
      v = v > 0.0f ? v : 0.0f;                 // ReLU
      atomicAdd(&agg[(long)sDst[e] * in_ch + f], v);
    }
  }
}

// ---------------------------------------------------------------------------
// Fragment packing helpers. Packed layouts (f16, 16 halves per lane):
//   A (activations, [M,K]):  idx = ((mtile*KT + kt)*32 + lane)*16 + j
//     row = mtile*16 + (lane&15)
//     k   = kt*32 + (j>>3)*16 + (lane>>4)*8 + (j&7)
//   B (weights, [K,Nout]):   idx = ((ntile*KT + kt)*32 + lane)*16 + j
//     k   = kt*32 + (j>>3)*16 + (lane>>4)*8 + (j&7)
//     col = ntile*16 + (lane&15)

// pack h = (1+eps)*x + agg straight into packed-A f16 (fuses the combine pass)
__global__ void gine_pack_h(const float* __restrict__ x,
                            const float* __restrict__ agg,
                            const float* __restrict__ epsP,
                            _Float16*    __restrict__ out,
                            int KT, int K, long total) {
  long gid = (long)blockIdx.x * blockDim.x + threadIdx.x;
  if (gid >= total) return;
  const float c1 = 1.0f + epsP[0];
  int  j    = (int)(gid & 15);
  int  lane = (int)((gid >> 4) & 31);
  long t    = gid >> 9;
  int  kt   = (int)(t % KT);
  long mt   = t / KT;
  long row  = mt * 16 + (lane & 15);
  int  k    = kt * 32 + ((j >> 3) << 4) + ((lane >> 4) << 3) + (j & 7);
  long src  = row * K + k;
  out[gid] = (_Float16)(c1 * x[src] + agg[src]);
}

// pack plain f32 activations into packed-A f16
__global__ void gine_pack_a(const float* __restrict__ X,
                            _Float16*    __restrict__ out,
                            int KT, int K, long total) {
  long gid = (long)blockIdx.x * blockDim.x + threadIdx.x;
  if (gid >= total) return;
  int  j    = (int)(gid & 15);
  int  lane = (int)((gid >> 4) & 31);
  long t    = gid >> 9;
  int  kt   = (int)(t % KT);
  long mt   = t / KT;
  long row  = mt * 16 + (lane & 15);
  int  k    = kt * 32 + ((j >> 3) << 4) + ((lane >> 4) << 3) + (j & 7);
  out[gid] = (_Float16)X[row * K + k];
}

// pack weight [K, Nout] into packed-B f16
__global__ void gine_pack_w(const float* __restrict__ W,
                            _Float16*    __restrict__ out,
                            int KT, int Nout, long total) {
  long gid = (long)blockIdx.x * blockDim.x + threadIdx.x;
  if (gid >= total) return;
  int  j    = (int)(gid & 15);
  int  lane = (int)((gid >> 4) & 31);
  long t    = gid >> 9;
  int  kt   = (int)(t % KT);
  long nt   = t / KT;
  int  k    = kt * 32 + ((j >> 3) << 4) + ((lane >> 4) << 3) + (j & 7);
  long col  = nt * 16 + (lane & 15);
  out[gid] = (_Float16)W[(long)k * Nout + col];
}

// ---------------------------------------------------------------------------
// WMMA GEMM: C[M,Nout] = act(Apk @ Bpk + bias)
// 8 waves/block; each wave owns a 16(M) x 64(N) strip: 4 f32 accumulators
// share one A fragment per K-step of 32. Weights (512 KB packed) stay in L2.
__global__ void __launch_bounds__(256)
gine_gemm_wmma(const _Float16* __restrict__ Apk,
               const _Float16* __restrict__ Bpk,
               const float*    __restrict__ bias,
               float*          __restrict__ C,
               int Mtiles, int KT, int Nout, int act, float slope) {
  const int wave = threadIdx.x >> 5;
  const int lane = threadIdx.x & 31;
  const int mtile = blockIdx.x * 8 + wave;
  if (mtile >= Mtiles) return;                  // wave-uniform: EXEC stays all-1
  const int nt0 = blockIdx.y * 4;

  const v16h* __restrict__ A  = (const v16h*)Apk + (long)mtile * KT * 32 + lane;
  const v16h* __restrict__ B0 = (const v16h*)Bpk + (long)(nt0 + 0) * KT * 32 + lane;
  const v16h* __restrict__ B1 = (const v16h*)Bpk + (long)(nt0 + 1) * KT * 32 + lane;
  const v16h* __restrict__ B2 = (const v16h*)Bpk + (long)(nt0 + 2) * KT * 32 + lane;
  const v16h* __restrict__ B3 = (const v16h*)Bpk + (long)(nt0 + 3) * KT * 32 + lane;

  v8f acc0 = {}, acc1 = {}, acc2 = {}, acc3 = {};

  for (int kt = 0; kt < KT; ++kt) {
    const long o = (long)kt * 32;
    if (kt + 1 < KT)                            // hint next A fragment into L0
      __builtin_prefetch((const void*)(A + o + 32), 0, 3);
    v16h a  = A[o];
    v16h b0 = B0[o];
    v16h b1 = B1[o];
    v16h b2 = B2[o];
    v16h b3 = B3[o];
    acc0 = __builtin_amdgcn_wmma_f32_16x16x32_f16(false, a, false, b0, (short)0, acc0, false, false);
    acc1 = __builtin_amdgcn_wmma_f32_16x16x32_f16(false, a, false, b1, (short)0, acc1, false, false);
    acc2 = __builtin_amdgcn_wmma_f32_16x16x32_f16(false, a, false, b2, (short)0, acc2, false, false);
    acc3 = __builtin_amdgcn_wmma_f32_16x16x32_f16(false, a, false, b3, (short)0, acc3, false, false);
  }

  // C/D fragment: lane 0-15 -> rows 0..7 (VGPR r), lane 16-31 -> rows 8..15
  const int col   = lane & 15;
  const int rbase = (lane >> 4) << 3;
  const long rowg = (long)mtile * 16 + rbase;

#define GINE_STORE(ACC, T)                                                \
  {                                                                       \
    const int  cg = (nt0 + (T)) * 16 + col;                               \
    const float bb = bias[cg];                                            \
    _Pragma("unroll")                                                     \
    for (int r = 0; r < 8; ++r) {                                         \
      float v = (ACC)[r] + bb;                                            \
      if (act) v = v > 0.0f ? v : v * slope;                              \
      C[(rowg + r) * Nout + cg] = v;                                      \
    }                                                                     \
  }
  GINE_STORE(acc0, 0)
  GINE_STORE(acc1, 1)
  GINE_STORE(acc2, 2)
  GINE_STORE(acc3, 3)
#undef GINE_STORE
}

// ---------------------------------------------------------------------------
// Final head: out[n] = dot(x[n,:], fw) + fb   (wave32 shuffle reduction)
__global__ void __launch_bounds__(256)
gine_final_dot(const float* __restrict__ x,
               const float* __restrict__ fw,
               const float* __restrict__ fb,
               float* __restrict__ out, int Nn, int K) {
  const int wave = threadIdx.x >> 5;
  const int lane = threadIdx.x & 31;
  const int node = blockIdx.x * 8 + wave;
  if (node >= Nn) return;
  float s = 0.0f;
  for (int f = lane; f < K; f += 32)
    s = fmaf(x[(long)node * K + f], fw[f], s);
#pragma unroll
  for (int o = 16; o > 0; o >>= 1) s += __shfl_xor(s, o, 32);
  if (lane == 0) out[node] = s + fb[0];
}

// ---------------------------------------------------------------------------
extern "C" void kernel_launch(void* const* d_in, const int* in_sizes, int n_in,
                              void* d_out, int out_size, void* d_ws, size_t ws_size,
                              hipStream_t stream) {
  (void)in_sizes; (void)n_in; (void)out_size; (void)ws_size;

  const float* x0 = (const float*)d_in[0];      // [N,128]
  const int*   ei = (const int*)  d_in[1];      // [2,E]
  const float* ea = (const float*)d_in[2];      // [E,16]
  const float* fw = (const float*)d_in[24];     // [512,1]
  const float* fb = (const float*)d_in[25];     // [1]

  const int N = GN_N, E = GN_E, H = GN_H;
  const int Mtiles = N / 16;                    // 3125 exactly

  // Workspace layout (stream-ordered buffer reuse):
  //   xbuf  f32 [N,512]  : layer output x
  //   tbuf  f32 [N,512]  : agg, then GEMM1 output t (reused)
  //   pk    f16 [N,512]  : packed-A fragments (h, then t; reused)
  //   wpk   f16 [512,512]: packed-B fragments (w1 then w2; reused)
  float*    xbuf = (float*)d_ws;
  float*    tbuf = xbuf + (size_t)N * H;
  _Float16* pk   = (_Float16*)(tbuf + (size_t)N * H);
  _Float16* wpk  = pk + (size_t)N * H;

  const float* xcur = x0;
  int in_ch = 128;

  for (int l = 0; l < 3; ++l) {
    const float* eps = (const float*)d_in[3 + 7 * l + 0];
    const float* elw = (const float*)d_in[3 + 7 * l + 1];
    const float* elb = (const float*)d_in[3 + 7 * l + 2];
    const float* w1  = (const float*)d_in[3 + 7 * l + 3];
    const float* b1  = (const float*)d_in[3 + 7 * l + 4];
    const float* w2  = (const float*)d_in[3 + 7 * l + 5];
    const float* b2  = (const float*)d_in[3 + 7 * l + 6];
    const int KT_in = in_ch / 32;

    // 1) agg = 0
    const long nagg = (long)N * in_ch;
    gine_zero_f32<<<(int)((nagg + 255) / 256), 256, 0, stream>>>(tbuf, nagg);

    // 2) scatter: agg[dst] += relu(x[src] + edge_attr@elw + elb)
    gine_edge_fused<<<(E + EDGE_TILE - 1) / EDGE_TILE, 256, 0, stream>>>(
        xcur, ei, ea, elw, elb, tbuf, in_ch, E);

    // 3) pack h = (1+eps)*x + agg -> f16 A-fragments
    const long totA = (long)N * in_ch;
    gine_pack_h<<<(int)((totA + 255) / 256), 256, 0, stream>>>(
        xcur, tbuf, eps, pk, KT_in, in_ch, totA);

    // 4) GEMM1: t = leaky(h @ w1 + b1)   [N,in_ch]@[in_ch,512]
    const long totW1 = (long)in_ch * H;
    gine_pack_w<<<(int)((totW1 + 255) / 256), 256, 0, stream>>>(
        w1, wpk, KT_in, H, totW1);
    {
      dim3 grid((Mtiles + 7) / 8, H / 64);
      gine_gemm_wmma<<<grid, 256, 0, stream>>>(pk, wpk, b1, tbuf,
                                               Mtiles, KT_in, H, 1, GN_SLOPE);
    }

    // 5) GEMM2: x = t @ w2 + b2 (+leaky except last layer)
    const long totA2 = (long)N * H;
    gine_pack_a<<<(int)((totA2 + 255) / 256), 256, 0, stream>>>(
        tbuf, pk, H / 32, H, totA2);
    const long totW2 = (long)H * H;
    gine_pack_w<<<(int)((totW2 + 255) / 256), 256, 0, stream>>>(
        w2, wpk, H / 32, H, totW2);
    {
      dim3 grid((Mtiles + 7) / 8, H / 64);
      gine_gemm_wmma<<<grid, 256, 0, stream>>>(pk, wpk, b2, xbuf,
                                               Mtiles, H / 32, H,
                                               (l < 2) ? 1 : 0, GN_SLOPE);
    }

    xcur = xbuf;
    in_ch = H;
  }

  // Final: out[n] = x[n,:] . fw + fb
  gine_final_dot<<<(N + 7) / 8, 256, 0, stream>>>(xbuf, fw, fb,
                                                  (float*)d_out, N, H);
}
